// VecLeadingZeroDetector108_43860206027326
// MI455X (gfx1250) — matrix operand order repaired and checked by reading, not compiled, not run on hardware
//
#include <hip/hip_runtime.h>
#include <cstdint>

typedef __attribute__((ext_vector_type(16))) _Float16 v16h;
typedef __attribute__((ext_vector_type(8)))  float    v8f;
typedef __attribute__((ext_vector_type(4)))  float    f32x4;

#define NBITS 108

// POS_EXT[i][j] = (i >> (6-j)) & 1, i in [0,128), laid out as WMMA 16-bit B
// fragments (32x16 per K=32 chunk): lane column n = l%16, halves e=0..15 give
// K = (e>=8?16:0) + (l>=16?8:0) + (e&7). Columns 7..15 are zero. Built at compile time.
struct alignas(32) BTab { uint16_t h[4][32][16]; };

static constexpr BTab makeBTab() {
    BTab t{};
    for (int c = 0; c < 4; ++c)
        for (int l = 0; l < 32; ++l)
            for (int e = 0; e < 16; ++e) {
                const int n = l & 15;
                const int i = c * 32 + ((e >= 8) ? 16 : 0) + ((l >= 16) ? 8 : 0) + (e & 7);
                t.h[c][l][e] = (n < 7 && ((i >> (6 - n)) & 1)) ? 0x3C00 : 0x0000;
            }
    return t;
}
__constant__ BTab g_btab = makeBTab();

// One wave = 32 rows. Phase 1: arithmetic bit-pack of the row into a 108-bit
// mask (inputs are exactly 0.0/1.0, 24 bits per fp32 accumulator is exact),
// first-set index via ctz chain; bit 108 is pre-set so all-zero rows give
// idx=108 for free. Phase 2: two 16-row tiles of
// D = onehot(idx)[16x128] x POS_EXT[128x7] via v_wmma_f32_16x16x32_f16.
__global__ __launch_bounds__(256) void lzd108_wmma_kernel(const float* __restrict__ X,
                                                          float* __restrict__ out,
                                                          int nrows) {
    const int lane = threadIdx.x & 31;
    const int base = blockIdx.x * 256 + (threadIdx.x >> 5) * 32;
    if (base >= nrows) return;                       // wave-uniform, EXEC stays all-1s

    // ---------------- phase 1: streaming bit-pack, one row per lane ----------------
    const int    row  = base + lane;
    const int    lrow = (row < nrows) ? row : (nrows - 1);       // clamp tail loads
    const f32x4* p    = (const f32x4*)(X + (size_t)lrow * NBITS);

    float acc[5] = {0.f, 0.f, 0.f, 0.f, 0.f};        // 24 exact bits each
    #pragma unroll
    for (int q = 0; q < NBITS / 4; ++q) {            // 27 x global_load_b128 (NT)
        const f32x4 v = __builtin_nontemporal_load(p + q);
        #pragma unroll
        for (int j = 0; j < 4; ++j) {
            const int i = q * 4 + j;
            acc[i / 24] = __builtin_fmaf(v[j], (float)(1u << (i % 24)), acc[i / 24]);
        }
    }
    const uint32_t u0 = (uint32_t)acc[0], u1 = (uint32_t)acc[1], u2 = (uint32_t)acc[2],
                   u3 = (uint32_t)acc[3], u4 = (uint32_t)acc[4];
    const uint32_t w0 = u0 | (u1 << 24);             // bits   0..31
    const uint32_t w1 = (u1 >> 8) | (u2 << 16);      // bits  32..63
    const uint32_t w2 = (u2 >> 16) | (u3 << 8);      // bits  64..95
    const uint32_t w3 = u4 | 0x1000u;                // bits 96..107, +bit108 always set
    const int idx = (w0 != 0u) ?      __builtin_ctz(w0)
                  : (w1 != 0u) ? 32 + __builtin_ctz(w1)
                  : (w2 != 0u) ? 64 + __builtin_ctz(w2)
                  :              96 + __builtin_ctz(w3);   // all-zero row -> 108

    // ---------------- B fragments from constant table ----------------
    v16h bfrag[4];
    #pragma unroll
    for (int c = 0; c < 4; ++c)
        bfrag[c] = *(const v16h*)g_btab.h[c][lane];

    const int  n         = lane & 15;
    const int  mhi       = (lane >= 16) ? 8 : 0;
    const bool fullstore = (base + 32) <= nrows;     // wave-uniform

    // ---------------- phase 2: two 16-row WMMA tiles ----------------
    #pragma unroll
    for (int t = 0; t < 2; ++t) {
        // idx of this lane's A-row lives in lane t*16 + n; fold mhi out so every
        // one-hot compare uses a pure literal constant (2 VALU per element).
        const int rr = __shfl(idx, t * 16 + n, 32) - mhi;

        v8f d = {0.f, 0.f, 0.f, 0.f, 0.f, 0.f, 0.f, 0.f};
        #pragma unroll
        for (int c = 0; c < 4; ++c) {
            // 16-bit A (16x32): element e covers K = (e>=8?16:0) + mhi + (e&7)
            v16h a;
            #pragma unroll
            for (int e = 0; e < 16; ++e) {
                const int kc = c * 32 + ((e >= 8) ? 16 : 0) + (e & 7);
                a[e] = (_Float16)((rr == kc) ? 1.0f : 0.0f);
            }
            d = __builtin_amdgcn_wmma_f32_16x16x32_f16(false, a, false, bfrag[c],
                                                       (short)0, d, false, false);
        }

        // C/D layout: lane column n = l%16, VGPR k is row m = k + mhi
        if (n < 7) {
            const int rbase = base + t * 16 + mhi;
            if (fullstore) {
                #pragma unroll
                for (int k = 0; k < 8; ++k)
                    __builtin_nontemporal_store(d[k], &out[(size_t)(rbase + k) * 7 + n]);
            } else {
                #pragma unroll
                for (int k = 0; k < 8; ++k)
                    if (rbase + k < nrows)
                        out[(size_t)(rbase + k) * 7 + n] = d[k];
            }
        }
    }
}

extern "C" void kernel_launch(void* const* d_in, const int* in_sizes, int n_in,
                              void* d_out, int out_size, void* d_ws, size_t ws_size,
                              hipStream_t stream) {
    const float* X   = (const float*)d_in[0];
    float*       out = (float*)d_out;
    const int nrows  = in_sizes[0] / NBITS;          // 2,000,000
    const int blocks = (nrows + 255) / 256;          // 256 rows per block (8 waves x 32)
    lzd108_wmma_kernel<<<blocks, 256, 0, stream>>>(X, out, nrows);
}
